// MoELoRALinear_39943195853509
// MI455X (gfx1250) — matrix operand order repaired and checked by reading, not compiled, not run on hardware
//
#include <hip/hip_runtime.h>

// ---------------------------------------------------------------------------
// MoE-LoRA linear for MI455X (gfx1250), bf16 WMMA path.
//   out = x @ W^T + 2.0 * sum_e w_e * (x @ A_e) @ B_e
// Main kernel: LDS double-buffered GEMM using gfx1250 async global->LDS
// copies (ASYNCcnt) feeding v_wmma_f32_16x16x32_bf16, with the LoRA rank-128
// update fused as a K-tail on the same accumulators (single output write).
// ---------------------------------------------------------------------------

typedef __attribute__((ext_vector_type(16))) __bf16 v16bf;
typedef __attribute__((ext_vector_type(8)))  __bf16 v8bf;
typedef __attribute__((ext_vector_type(8)))  float  v8f;

#define D_DIM 4096
#define M_TOK 8192
#define E_NUM 8
#define R_NUM 16
#define ER_DIM 128
#define LORA_SCALE 2.0f

#define KSTEP 64
#define AS_STRIDE 80   // 64 k + 16 pad (elems); row stride 160B (32B multiple)
#define BS_STRIDE 144  // 128 n + 16 pad (elems); row stride 288B (32B multiple)

// --- CDNA5 primitives ------------------------------------------------------

static __device__ __forceinline__ v8f wmma_bf16(v16bf a, v16bf b, v8f c) {
  // 8 args: (neg_a, A, neg_b, B, c_mod, C, reuse_a, reuse_b)
  return __builtin_amdgcn_wmma_f32_16x16x32_bf16(
      false, a, false, b, (short)0, c, false, false);
}

// Async global -> LDS copy, 16B per lane, tracked by ASYNCcnt.
static __device__ __forceinline__ void async_g2l_b128(uint32_t lds_addr,
                                                      uint64_t gaddr) {
  asm volatile("global_load_async_to_lds_b128 %0, %1, off"
               :: "v"(lds_addr), "v"(gaddr)
               : "memory");
}

static __device__ __forceinline__ void wait_async0() {
#if __has_builtin(__builtin_amdgcn_s_wait_asynccnt)
  __builtin_amdgcn_s_wait_asynccnt(0);
#else
  asm volatile("s_wait_asynccnt 0" ::: "memory");
#endif
}

// Low 32 bits of a flat LDS pointer == LDS byte address (ISA 10.2 aperture).
static __device__ __forceinline__ uint32_t lds_addr32(const void* p) {
  return (uint32_t)(uintptr_t)p;
}

// A-matrix 16x32 bf16 fragment per ISA layout:
//   lane L: row m = m0 + (L&15); half h = L>>4
//   VGPR0..3 hold K = h*8 + 0..7 ; VGPR4..7 hold K = 16 + h*8 + 0..7
static __device__ __forceinline__ v16bf make_a(v8bf lo, v8bf hi) {
  return __builtin_shufflevector(lo, hi, 0, 1, 2, 3, 4, 5, 6, 7,
                                 8, 9, 10, 11, 12, 13, 14, 15);
}

static __device__ __forceinline__ v16bf load_a_frag(const __bf16* A, int lda,
                                                    int m0, int k0, int lane) {
  int l = lane & 15, h = lane >> 4;
  const __bf16* p = A + (size_t)(m0 + l) * lda + (k0 + h * 8);
  return make_a(*(const v8bf*)p, *(const v8bf*)(p + 16));
}

// --- Kernel 1: x -> bf16, router logits + softmax + top-2 weights ----------

__global__ void k_x_router(const float* __restrict__ x,
                           const float* __restrict__ routerW,
                           __bf16* __restrict__ xbf,
                           float* __restrict__ w) {
  __shared__ float red[E_NUM][256];
  int tok = blockIdx.x;
  int tid = threadIdx.x;
  float acc[E_NUM];
#pragma unroll
  for (int e = 0; e < E_NUM; ++e) acc[e] = 0.0f;

  const float* xr = x + (size_t)tok * D_DIM;
#pragma unroll 4
  for (int it = 0; it < D_DIM / 256; ++it) {
    int i = tid + it * 256;
    float v = xr[i];
    xbf[(size_t)tok * D_DIM + i] = (__bf16)v;
#pragma unroll
    for (int e = 0; e < E_NUM; ++e) acc[e] += v * routerW[(size_t)e * D_DIM + i];
  }
#pragma unroll
  for (int e = 0; e < E_NUM; ++e) red[e][tid] = acc[e];
  __syncthreads();
  for (int off = 128; off > 0; off >>= 1) {
    if (tid < off) {
#pragma unroll
      for (int e = 0; e < E_NUM; ++e) red[e][tid] += red[e][tid + off];
    }
    __syncthreads();
  }
  if (tid == 0) {
    float logit[E_NUM], prob[E_NUM];
    float mx = -3.4e38f;
#pragma unroll
    for (int e = 0; e < E_NUM; ++e) { logit[e] = red[e][0]; mx = fmaxf(mx, logit[e]); }
    float s = 0.0f;
#pragma unroll
    for (int e = 0; e < E_NUM; ++e) { prob[e] = __expf(logit[e] - mx); s += prob[e]; }
    float inv = 1.0f / s;
#pragma unroll
    for (int e = 0; e < E_NUM; ++e) prob[e] *= inv;
    int i1 = 0; float v1 = prob[0];
#pragma unroll
    for (int e = 1; e < E_NUM; ++e) if (prob[e] > v1) { v1 = prob[e]; i1 = e; }
    int i2 = (i1 == 0) ? 1 : 0; float v2 = prob[i2];
#pragma unroll
    for (int e = 0; e < E_NUM; ++e)
      if (e != i1 && prob[e] > v2) { v2 = prob[e]; i2 = e; }
    float rinv = 1.0f / (v1 + v2);
#pragma unroll
    for (int e = 0; e < E_NUM; ++e) {
      float we = (e == i1) ? v1 * rinv : (e == i2) ? v2 * rinv : 0.0f;
      w[(size_t)tok * E_NUM + e] = we;
    }
  }
}

// --- Kernel 2: W [O][D] fp32 -> WT [D][O] bf16 (tiled transpose) -----------

__global__ void k_wt(const float* __restrict__ Wf, __bf16* __restrict__ WT) {
  __shared__ float t[32][33];
  int tx = threadIdx.x, ty = threadIdx.y;
  int o0 = blockIdx.x * 32, d0 = blockIdx.y * 32;
#pragma unroll
  for (int i = 0; i < 4; ++i)
    t[ty + i * 8][tx] = Wf[(size_t)(o0 + ty + i * 8) * D_DIM + d0 + tx];
  __syncthreads();
#pragma unroll
  for (int i = 0; i < 4; ++i)
    WT[(size_t)(d0 + ty + i * 8) * D_DIM + o0 + tx] = (__bf16)t[tx][ty + i * 8];
}

// --- Kernel 3: fp32 -> bf16 elementwise ------------------------------------

__global__ void k_cvt(const float* __restrict__ src, __bf16* __restrict__ dst,
                      int n) {
  int i = blockIdx.x * blockDim.x + threadIdx.x;
  if (i < n) dst[i] = (__bf16)src[i];
}

// --- Kernel 4: midw = (x_bf @ A_e) * w_e * 2.0 -> bf16 [8192 x 128] --------

__global__ void __launch_bounds__(256)
k_mid(const __bf16* __restrict__ xbf, const __bf16* __restrict__ Abf,
      const float* __restrict__ w, __bf16* __restrict__ midw) {
  int tid = threadIdx.x;
  int wave = tid >> 5, lane = tid & 31;
  int m0 = blockIdx.x * 128 + wave * 16;

  v8f acc[E_NUM] = {};
  for (int k0 = 0; k0 < D_DIM; k0 += 32) {
    v16bf a = load_a_frag(xbf, D_DIM, m0, k0, lane);
#pragma unroll
    for (int e = 0; e < E_NUM; ++e) {
      const __bf16* bp = Abf + ((size_t)e * D_DIM + k0 + (lane >> 4) * 8) * R_NUM + (lane & 15);
      v16bf b;
      for (int s = 0; s < 16; ++s) b[s] = bp[(s + (s & 8)) * R_NUM];
      acc[e] = wmma_bf16(a, b, acc[e]);
    }
  }
  int l = lane & 15, h = lane >> 4;
#pragma unroll
  for (int e = 0; e < E_NUM; ++e) {
#pragma unroll
    for (int v = 0; v < 8; ++v) {
      int m = m0 + v + h * 8;
      float val = acc[e][v] * w[(size_t)m * E_NUM + e] * LORA_SCALE;
      midw[(size_t)m * ER_DIM + e * R_NUM + l] = (__bf16)val;
    }
  }
}

// --- Kernel 5: out = x_bf @ WT + midw @ loraB (fused K = 4096 + 128) -------
// LDS double-buffered main loop using async global->LDS (ASYNCcnt), then a
// register-direct LoRA K-tail (operands are L2-resident: 2MB + 1MB).

__global__ void __launch_bounds__(256)
k_main(const __bf16* __restrict__ xbf, const __bf16* __restrict__ WT,
       const __bf16* __restrict__ midw, const __bf16* __restrict__ Bbf,
       float* __restrict__ out) {
  __shared__ __align__(32) __bf16 As[2][128 * AS_STRIDE];
  __shared__ __align__(32) __bf16 Bs[2][KSTEP * BS_STRIDE];

  const int tid = threadIdx.x;
  const int wave = tid >> 5, lane = tid & 31;
  const int l = lane & 15, h = lane >> 4;
  const int m_base = blockIdx.y * 128;
  const int n0 = blockIdx.x * 128;

  const uint32_t as0 = lds_addr32(&As[0][0]);
  const uint32_t bs0 = lds_addr32(&Bs[0][0]);

  v8f acc[8] = {};

  // Cooperative async stage of one K-slab: A 128x64, B 64x128 (16KB each).
  auto stage = [&](int buf, int k0) {
    uint32_t abase = as0 + (uint32_t)buf * (128 * AS_STRIDE * 2);
    uint32_t bbase = bs0 + (uint32_t)buf * (KSTEP * BS_STRIDE * 2);
#pragma unroll
    for (int t = 0; t < 4; ++t) {       // A tile: 1024 chunks of 8 bf16
      int c = tid + t * 256;
      int row = c >> 3, sub = c & 7;
      uint64_t g = (uint64_t)(uintptr_t)(xbf + (size_t)(m_base + row) * D_DIM +
                                         k0 + sub * 8);
      async_g2l_b128(abase + (uint32_t)(row * AS_STRIDE + sub * 8) * 2, g);
    }
#pragma unroll
    for (int t = 0; t < 4; ++t) {       // B tile: 1024 chunks of 8 bf16
      int c = tid + t * 256;
      int kr = c >> 4, sub = c & 15;
      uint64_t g = (uint64_t)(uintptr_t)(WT + (size_t)(k0 + kr) * D_DIM +
                                         n0 + sub * 8);
      async_g2l_b128(bbase + (uint32_t)(kr * BS_STRIDE + sub * 8) * 2, g);
    }
  };

  stage(0, 0);
  for (int kt = 0; kt < D_DIM / KSTEP; ++kt) {
    wait_async0();        // my buf(kt) copies have landed in LDS
    __syncthreads();      // everyone's copies landed; prev reads done
    if (kt + 1 < D_DIM / KSTEP) stage((kt + 1) & 1, (kt + 1) * KSTEP);

    const __bf16* a_t = As[kt & 1];
    const __bf16* b_t = Bs[kt & 1];
#pragma unroll
    for (int j = 0; j < KSTEP / 32; ++j) {
      const __bf16* ap = a_t + (wave * 16 + l) * AS_STRIDE + j * 32 + h * 8;
      v16bf a = make_a(*(const v8bf*)ap, *(const v8bf*)(ap + 16));
      const __bf16* brow = b_t + (j * 32 + h * 8) * BS_STRIDE + l;
#pragma unroll
      for (int t = 0; t < 8; ++t) {
        v16bf b;
        for (int s = 0; s < 16; ++s) b[s] = brow[(s + (s & 8)) * BS_STRIDE + t * 16];
        acc[t] = wmma_bf16(a, b, acc[t]);
      }
    }
  }

  // LoRA tail: K = 128, scale & routing weights pre-folded into midw.
  const int m0 = m_base + wave * 16;
#pragma unroll
  for (int k0 = 0; k0 < ER_DIM; k0 += 32) {
    v16bf a = load_a_frag(midw, ER_DIM, m0, k0, lane);
    const __bf16* brow = Bbf + (size_t)(k0 + h * 8) * D_DIM + n0 + l;
#pragma unroll
    for (int t = 0; t < 8; ++t) {
      v16bf b;
      for (int s = 0; s < 16; ++s) b[s] = brow[(size_t)(s + (s & 8)) * D_DIM + t * 16];
      acc[t] = wmma_bf16(a, b, acc[t]);
    }
  }

  // Epilogue: C/D layout -> lane holds (m = v + h*8, n = l) per VGPR v.
#pragma unroll
  for (int t = 0; t < 8; ++t) {
#pragma unroll
    for (int v = 0; v < 8; ++v) {
      int m = m0 + v + h * 8;
      out[(size_t)m * D_DIM + n0 + t * 16 + l] = acc[t][v];
    }
  }
}

// --- Launch ----------------------------------------------------------------

extern "C" void kernel_launch(void* const* d_in, const int* in_sizes, int n_in,
                              void* d_out, int out_size, void* d_ws, size_t ws_size,
                              hipStream_t stream) {
  const float* x       = (const float*)d_in[0];  // [4,2048,4096]
  const float* base_W  = (const float*)d_in[1];  // [4096,4096]
  const float* routerW = (const float*)d_in[2];  // [8,4096]
  const float* lora_A  = (const float*)d_in[3];  // [8,4096,16]
  const float* lora_B  = (const float*)d_in[4];  // [8,16,4096]
  float* out = (float*)d_out;                    // [4,2048,4096]

  char* ws = (char*)d_ws;
  size_t off = 0;
  __bf16* xbf  = (__bf16*)(ws + off); off += (size_t)M_TOK * D_DIM * 2;   // 64 MiB
  __bf16* WT   = (__bf16*)(ws + off); off += (size_t)D_DIM * D_DIM * 2;   // 32 MiB
  __bf16* Abf  = (__bf16*)(ws + off); off += (size_t)E_NUM * D_DIM * R_NUM * 2;
  __bf16* Bbf  = (__bf16*)(ws + off); off += (size_t)E_NUM * R_NUM * D_DIM * 2;
  float*  wrt  = (float*)(ws + off);  off += (size_t)M_TOK * E_NUM * 4;
  __bf16* midw = (__bf16*)(ws + off); off += (size_t)M_TOK * ER_DIM * 2;
  (void)off; (void)ws_size; (void)in_sizes; (void)n_in; (void)out_size;

  // 1) convert x + router weights
  k_x_router<<<M_TOK, 256, 0, stream>>>(x, routerW, xbf, wrt);
  // 2) transpose/convert base_W -> WT bf16 [D][O]
  k_wt<<<dim3(D_DIM / 32, D_DIM / 32), dim3(32, 8), 0, stream>>>(base_W, WT);
  // 3) convert LoRA weights to bf16
  {
    int nA = E_NUM * D_DIM * R_NUM;
    k_cvt<<<(nA + 255) / 256, 256, 0, stream>>>(lora_A, Abf, nA);
    int nB = E_NUM * R_NUM * D_DIM;
    k_cvt<<<(nB + 255) / 256, 256, 0, stream>>>(lora_B, Bbf, nB);
  }
  // 4) midw = (x @ A_e) * w_e * SCALE
  k_mid<<<M_TOK / 128, 256, 0, stream>>>(xbf, Abf, wrt, midw);
  // 5) out = x @ W^T + midw @ lora_B  (fused)
  k_main<<<dim3(D_DIM / 128, M_TOK / 128), 256, 0, stream>>>(xbf, WT, midw,
                                                             Bbf, out);
}
